// BinaryConv2D_79577154060629
// MI455X (gfx1250) — 8-bit matrix operand order repaired and checked by reading, compiled, not run on hardware
//
#include <hip/hip_runtime.h>


typedef __attribute__((ext_vector_type(16))) int          v16i;
typedef __attribute__((ext_vector_type(8)))  float        v8f;
typedef __attribute__((ext_vector_type(4)))  unsigned int v4u;
typedef int v4i_gcc __attribute__((vector_size(16)));

#define NIMG 32
#define HH   56
#define WW   56
#define CC   256
#define LDSX 66
#define PITCH 288
#define LROWB (LDSX * PITCH)
#define LDSB  (3 * LROWB)
#define IN_FP8_BYTES (NIMG * HH * WW * CC)
#define WPACK_BYTES  (9 * 2 * 16 * 2048)

#if defined(__gfx1250__) &&                                        \
    __has_builtin(__builtin_amdgcn_global_load_async_to_lds_b128) && \
    __has_builtin(__builtin_amdgcn_s_wait_asynccnt)
#define USE_ASYNC_LDS 1
#else
#define USE_ASYNC_LDS 0
#endif

__device__ __forceinline__ unsigned int fp8_sign(float x) {
    return x > 0.0f ? 0x38u : (x < 0.0f ? 0xB8u : 0x00u);
}

__global__ void bin_input_kernel(const float* __restrict__ in,
                                 unsigned char* __restrict__ out) {
    int t   = blockIdx.x * 256 + threadIdx.x;
    int blk = t >> 3;
    int sub = t & 7;
    int hi  = sub >> 2;
    int g0  = (sub & 3) * 2;
    const float* src = in + (size_t)blk * 128 + g0 * 16 + hi * 8;
    float4 f[4];
    f[0] = *(const float4*)(src);
    f[1] = *(const float4*)(src + 4);
    f[2] = *(const float4*)(src + 16);
    f[3] = *(const float4*)(src + 20);
    unsigned int d[4];
#pragma unroll
    for (int j = 0; j < 4; ++j)
        d[j] = fp8_sign(f[j].x) | (fp8_sign(f[j].y) << 8) |
               (fp8_sign(f[j].z) << 16) | (fp8_sign(f[j].w) << 24);
    v4u v = { d[0], d[1], d[2], d[3] };
    *(v4u*)(out + (size_t)t * 16) = v;
}

__global__ void pack_weights_kernel(const float* __restrict__ wsrc,
                                    unsigned char* __restrict__ wdst) {
    int t    = blockIdx.x * 256 + threadIdx.x;
    int frag = t >> 9;
    int dw   = t & 511;
    int lane = dw >> 4;
    int b0   = (dw & 15) * 4;
    int tap  = frag >> 5;
    int rem  = frag & 31;
    int ck   = rem >> 4;
    int coT  = rem & 15;
    int hiL  = lane >> 4;
    int co   = coT * 16 + (lane & 15);
    unsigned int dword = 0;
#pragma unroll
    for (int j = 0; j < 4; ++j) {
        int b   = b0 + j;
        int K   = (b >> 4) * 32 + ((b >> 3) & 1) * 16 + hiL * 8 + (b & 7);
        int ci  = ck * 128 + K;
        float x = wsrc[((size_t)tap * CC + ci) * CC + co];
        dword |= fp8_sign(x) << (8 * j);
    }
    ((unsigned int*)wdst)[t] = dword;
}

__launch_bounds__(256)
__global__ void bconv_wmma_kernel(const unsigned char* __restrict__ binIn,
                                  const unsigned char* __restrict__ wpack,
                                  float* __restrict__ out) {
    __shared__ __align__(16) unsigned char lds[LDSB];

    const int y = blockIdx.x % HH;
    const int n = blockIdx.x / HH;

    for (int chunk = threadIdx.x; chunk < 3 * LDSX * 16; chunk += 256) {
        int r   = chunk / (LDSX * 16);
        int rem = chunk - r * (LDSX * 16);
        int xl  = rem >> 4;
        int c   = (rem & 15) << 4;
        int off = r * LROWB + xl * PITCH + c;
        int gy  = y + r - 1;
        int gx  = xl - 1;
        if ((unsigned)gy < (unsigned)HH && (unsigned)gx < (unsigned)WW) {
            const unsigned char* gp = binIn +
                (((size_t)(n * HH + gy) * WW + gx) * CC + c);
#if USE_ASYNC_LDS
            __builtin_amdgcn_global_load_async_to_lds_b128(
                (__attribute__((address_space(1))) v4i_gcc*)gp,
                (__attribute__((address_space(3))) v4i_gcc*)(lds + off),
                0, 0);
#else
            *(v4u*)(lds + off) = *(const v4u*)gp;
#endif
        } else {
            v4u z = { 0u, 0u, 0u, 0u };
            *(v4u*)(lds + off) = z;
        }
    }
#if USE_ASYNC_LDS
    __builtin_amdgcn_s_wait_asynccnt(0);
#endif
    __syncthreads();

    const int w    = threadIdx.x >> 5;
    const int lane = threadIdx.x & 31;
    const int mlo  = lane & 15;
    const int hi   = lane >> 4;

    v8f acc[2][4];
#pragma unroll
    for (int nt = 0; nt < 2; ++nt)
#pragma unroll
        for (int mt = 0; mt < 4; ++mt) acc[nt][mt] = (v8f)0.f;

    for (int tap = 0; tap < 9; ++tap) {
        const int kh = tap / 3, kw = tap % 3;
#pragma unroll
        for (int ck = 0; ck < 2; ++ck) {
            union { v16i v; v4u q[4]; } B0, B1;
            {
                const unsigned char* fb = wpack +
                    ((size_t)((tap * 2 + ck) * 16)) * 2048 + lane * 64;
                const v4u* bp0 = (const v4u*)(fb + (size_t)w * 2048);
                const v4u* bp1 = (const v4u*)(fb + (size_t)(w + 8) * 2048);
#pragma unroll
                for (int j = 0; j < 4; ++j) { B0.q[j] = bp0[j]; B1.q[j] = bp1[j]; }
            }
#pragma unroll
            for (int mt = 0; mt < 4; ++mt) {
                int abase = kh * LROWB + (mt * 16 + mlo + kw) * PITCH
                          + ck * 128 + hi * 64;
                union { v16i v; v4u q[4]; } A;
#pragma unroll
                for (int j = 0; j < 4; ++j)
                    A.q[j] = *(const v4u*)(lds + abase + j * 16);

                acc[0][mt] = __builtin_amdgcn_wmma_f32_16x16x128_fp8_fp8(
                    A.v, B0.v, (short)0, acc[0][mt], false, false);
                acc[1][mt] = __builtin_amdgcn_wmma_f32_16x16x128_fp8_fp8(
                    A.v, B1.v, (short)0, acc[1][mt], false, false);
            }
        }
    }

    float* orow = out + ((size_t)(n * HH + y)) * WW * CC;
#pragma unroll
    for (int nt = 0; nt < 2; ++nt) {
        const int co = nt * 128 + w * 16 + mlo;
#pragma unroll
        for (int mt = 0; mt < 4; ++mt) {
            int xbase = mt * 16 + hi * 8;
            if (xbase < WW) {
#pragma unroll
                for (int i = 0; i < 8; ++i)
                    __builtin_nontemporal_store(acc[nt][mt][i],
                        orow + (size_t)(xbase + i) * CC + co);
            }
        }
    }
}

extern "C" void kernel_launch(void* const* d_in, const int* in_sizes, int n_in,
                              void* d_out, int out_size, void* d_ws, size_t ws_size,
                              hipStream_t stream) {
    const float* x = (const float*)d_in[0];
    const float* k = (const float*)d_in[1];
    float* out     = (float*)d_out;

    unsigned char* binIn = (unsigned char*)d_ws;
    unsigned char* wpack = binIn + IN_FP8_BYTES;

    bin_input_kernel<<<dim3(IN_FP8_BYTES / 16 / 256), dim3(256), 0, stream>>>(x, binIn);
    pack_weights_kernel<<<dim3(WPACK_BYTES / 4 / 256), dim3(256), 0, stream>>>(k, wpack);
    bconv_wmma_kernel<<<dim3(NIMG * HH), dim3(256), 0, stream>>>(binIn, wpack, out);
}
